// HGTModel_48223892800085
// MI455X (gfx1250) — compile-verified, matrix operand order checked by reading, not run on hardware
//
#include <hip/hip_runtime.h>
#include <math.h>

// ---------------------------------------------------------------------------
// CDNA5 (gfx1250) HGT forward.  wave32, WMMA bf16 for all 128-deep GEMMs.
// Weights converted f32->bf16 + pre-swizzled into CDNA5 B-fragment order ONCE
// (wprep); GEMM LDS staging is a single TDM tensor_load_to_lds (fallback:
// vector copy).  Activations kept bf16 between GEMMs; f32 for atomics/edges.
// ---------------------------------------------------------------------------

typedef __attribute__((ext_vector_type(16))) __bf16 v16bf;
typedef __attribute__((ext_vector_type(8)))  float  v8f;
typedef __attribute__((ext_vector_type(4)))  unsigned uv4;
typedef __attribute__((ext_vector_type(8)))  int      iv8;
typedef __attribute__((ext_vector_type(4)))  int      iv4;

#if defined(__has_builtin)
#if __has_builtin(__builtin_amdgcn_tensor_load_to_lds) && \
    __has_builtin(__builtin_amdgcn_s_wait_tensorcnt)
#define USE_TDM 1
#endif
#endif

static __device__ __forceinline__ unsigned short f2bf(float f) {
  union { float f; unsigned u; } c; c.f = f;
  unsigned u = c.u;
  u += 0x7fffu + ((u >> 16) & 1u);   // round-to-nearest-even
  return (unsigned short)(u >> 16);
}
static __device__ __forceinline__ float bf2f(unsigned short s) {
  union { unsigned u; float f; } c; c.u = ((unsigned)s) << 16;
  return c.f;
}
static __device__ __forceinline__ float gelu_exact(float x) {
  return 0.5f * x * (1.0f + erff(x * 0.70710678118654752f));
}

union AFrag { v16bf v; unsigned short s[16]; uint4 q[2]; };
union BFrag { v16bf v; uint4 q[2]; };

// ---------------------------------------------------------------------------
// One-shot weight prep: convert W (KIN x NOUT, f32, row-major) to bf16 in
// swizzled B-fragment order:
//   idx = kb*NOUT*32 + n*32 + hh*16 + j ; k = kb*32 + (j>>3)*16 + hh*8 + (j&7)
// ---------------------------------------------------------------------------
__global__ void wprep_kernel(const float* __restrict__ W,
                             unsigned short* __restrict__ out,
                             int KIN_, int NOUT_)
{
  const int idx = blockIdx.x * blockDim.x + threadIdx.x;
  if (idx >= KIN_ * NOUT_) return;
  const int j  = idx & 15;
  const int hh = (idx >> 4) & 1;
  const int n  = (idx >> 5) % NOUT_;
  const int kb = idx / (NOUT_ * 32);
  const int k  = kb * 32 + ((j >> 3) << 4) + hh * 8 + (j & 7);
  out[idx] = f2bf(W[(size_t)k * NOUT_ + n]);
}

// ---------------------------------------------------------------------------
// act( X[rows x KIN] @ W[KIN x NOUT] + bias ), optional gelu on input
// (ACTIN=2), optional sigmoid-gated residual vs bf16 res (RES).
// Compile-time: AB16 = A operand is bf16 (16B vector loads, no convert);
// WF32/WB16 = which outputs to write; HASBIAS = add bias.
// Block = 256 threads = 8 waves; wave owns a 16-row strip x all NOUT columns.
// Wsw is bf16, pre-swizzled -> one TDM DMA stages the whole block into LDS;
// each lane's B fragment is then 32 contiguous LDS bytes (2x ds_load_b128).
// ---------------------------------------------------------------------------
template <int KIN, int NOUT, int ACTIN, int ACTOUT, bool RES, bool AB16,
          bool WF32, bool WB16, bool HASBIAS>
__global__ __launch_bounds__(256) void gemm_wmma_kernel(
    const void* __restrict__ Xv, int ldX, int xoff,
    const unsigned short* __restrict__ Wsw, const float* __restrict__ bias,
    float* __restrict__ Yf, unsigned short* __restrict__ Yb,
    int ldY, int yoff,
    const unsigned short* __restrict__ res16, const float* __restrict__ skipS,
    int nRows)
{
  constexpr int NT = NOUT / 16;
  constexpr int KB = KIN / 32;
  __shared__ __align__(32) unsigned short Wl[KIN * NOUT];

  const int tid = threadIdx.x;
#if USE_TDM
  // Stage the pre-swizzled weight block with the Tensor Data Mover:
  // 1-D tile of n8 8-byte elements, one descriptor, issued by wave 0.
  if (tid < 32) {
    constexpr unsigned n8 = (unsigned)(KIN * NOUT) / 4;   // 8B elements
    const unsigned long long ga = (unsigned long long)Wsw;
    const unsigned ldsAddr =
        (unsigned)(unsigned long long)(__attribute__((address_space(3))) void*)Wl;
    uv4 g0;
    g0.x = 1u;                                            // count=1, user D#
    g0.y = ldsAddr;                                       // lds_addr
    g0.z = (unsigned)(ga & 0xffffffffull);                // global_addr[31:0]
    g0.w = (unsigned)((ga >> 32) & 0x01ffffffull) | (2u << 30); // [56:32]+type=2
    iv8 g1;
    g1[0] = (3 << 16);                  // wg_mask=0, data_size=3 (8B), no flags
    g1[1] = (int)((n8 & 0xffffu) << 16);          // tensor_dim0[15:0]
    g1[2] = (int)((n8 >> 16) | (1u << 16));       // tensor_dim0[31:16], dim1=1
    g1[3] = (int)((n8 & 0xffffu) << 16);          // dim1 hi=0, tile_dim0=n8
    g1[4] = 1;                                    // tile_dim1=1, tile_dim2=0
    g1[5] = (int)n8;                              // tensor_dim0_stride[31:0]
    g1[6] = 0;                                    // stride hi, dim1_stride lo
    g1[7] = 0;
    iv4 z4 = { 0, 0, 0, 0 };
#if __clang_major__ >= 23
    iv8 z8 = { 0, 0, 0, 0, 0, 0, 0, 0 };
    __builtin_amdgcn_tensor_load_to_lds(g0, g1, z4, z4, z8, 0);
#else
    __builtin_amdgcn_tensor_load_to_lds(g0, g1, z4, z4, 0);
#endif
    __builtin_amdgcn_s_wait_tensorcnt(0);
  }
#else
  {
    constexpr int NV = (KIN * NOUT) / 8;       // uint4 chunks
    const uint4* ws = (const uint4*)Wsw;
    uint4* wd = (uint4*)Wl;
    #pragma unroll 4
    for (int i = tid; i < NV; i += 256) wd[i] = ws[i];
  }
#endif
  __syncthreads();

  const int wave = tid >> 5;
  const int lane = tid & 31;
  const int m    = lane & 15;   // row-in-tile (A) / col-in-tile (B,D)
  const int hh   = lane >> 4;   // K-half select
  const int rowBase = blockIdx.x * 128 + wave * 16;
  const int row = rowBase + m;
  const int rc  = row < nRows ? row : (nRows - 1);   // clamp: keep EXEC full

  v8f acc[NT];
  #pragma unroll
  for (int c = 0; c < NT; ++c)
    #pragma unroll
    for (int r = 0; r < 8; ++r) acc[c][r] = 0.0f;

  for (int kb = 0; kb < KB; ++kb) {
    const int k0 = kb * 32 + hh * 8;
    AFrag a;
    if constexpr (AB16) {
      const unsigned short* xrow =
          (const unsigned short*)Xv + (size_t)rc * ldX + xoff;
      a.q[0] = *(const uint4*)(xrow + k0);        // 8 bf16, K-run 0
      a.q[1] = *(const uint4*)(xrow + k0 + 16);   // 8 bf16, K-run 1
    } else {
      const float* xrow = (const float*)Xv + (size_t)rc * ldX + xoff;
      float4 x0 = *(const float4*)(xrow + k0);
      float4 x1 = *(const float4*)(xrow + k0 + 4);
      float4 x2 = *(const float4*)(xrow + k0 + 16);
      float4 x3 = *(const float4*)(xrow + k0 + 20);
      float t[16] = { x0.x,x0.y,x0.z,x0.w, x1.x,x1.y,x1.z,x1.w,
                      x2.x,x2.y,x2.z,x2.w, x3.x,x3.y,x3.z,x3.w };
      #pragma unroll
      for (int j = 0; j < 16; ++j) {
        float v = t[j];
        if (ACTIN == 2) v = gelu_exact(v);
        a.s[j] = f2bf(v);
      }
    }
    #pragma unroll
    for (int c = 0; c < NT; ++c) {
      BFrag b;
      const uint4* bp =
        (const uint4*)&Wl[kb * NOUT * 32 + (c * 16 + m) * 32 + hh * 16];
      b.q[0] = bp[0];
      b.q[1] = bp[1];
      acc[c] = __builtin_amdgcn_wmma_f32_16x16x32_bf16(
          false, a.v, false, b.v, (short)0, acc[c], false, false);
    }
  }

  // ----------------------------- epilogue ---------------------------------
  float beta = 0.f, ombeta = 0.f;
  if constexpr (RES) {
    float s = *skipS; beta = 1.0f / (1.0f + expf(-s)); ombeta = 1.0f - beta;
  }

  float bv[NT];
  #pragma unroll
  for (int c = 0; c < NT; ++c)
    bv[c] = HASBIAS ? bias[c * 16 + m] : 0.0f;

  const int row0 = rowBase + hh * 8;          // first of this lane's 8 rows

  auto doRow = [&](int c, int r, size_t base) {
    float val = acc[c][r] + bv[c];
    if (ACTOUT == 1) val = val > 0.f ? val : 0.01f * val;
    const size_t o = base + (size_t)r * ldY;
    if constexpr (RES) val = beta * val + ombeta * bf2f(res16[o]);
    if constexpr (WF32) Yf[o] = val;
    if constexpr (WB16) Yb[o] = f2bf(val);
  };

  if (rowBase + 16 <= nRows) {                // uniform fast path: no guards
    #pragma unroll
    for (int c = 0; c < NT; ++c) {
      const size_t base = (size_t)row0 * ldY + yoff + c * 16 + m;
      #pragma unroll
      for (int r = 0; r < 8; ++r) doRow(c, r, base);
    }
  } else {                                    // last block: guard per row
    #pragma unroll
    for (int r = 0; r < 8; ++r) {
      if (row0 + r < nRows) {
        #pragma unroll
        for (int c = 0; c < NT; ++c) {
          const size_t base = (size_t)row0 * ldY + yoff + c * 16 + m;
          doRow(c, r, base);
        }
      }
    }
  }
}

// --------------------------- edge-phase kernels ----------------------------
// Order-preserving float<->uint encoding for atomicMax-based segment max.
static __device__ __forceinline__ unsigned fenc(float f) {
  union { float f; unsigned u; } c; c.f = f;
  return (c.u & 0x80000000u) ? ~c.u : (c.u | 0x80000000u);
}
static __device__ __forceinline__ float fdec(unsigned u) {
  union { unsigned u; float f; } c;
  c.u = (u & 0x80000000u) ? (u & 0x7fffffffu) : ~u;
  return c.f;
}

__global__ void edge_alpha_kernel(const int* __restrict__ ei, int E,
    const float* __restrict__ q, const float* __restrict__ kr,
    const float* __restrict__ pHead, float* __restrict__ alpha,
    unsigned int* __restrict__ amax)
{
  const int e = blockIdx.x * 8 + (threadIdx.x >> 5);
  if (e >= E) return;                       // uniform per wave
  const int lane = threadIdx.x & 31;
  const int src = ei[e];
  const int dst = ei[E + e];
  const float4 qv = *(const float4*)(q  + (size_t)dst * 128 + lane * 4);
  const float4 kv = *(const float4*)(kr + (size_t)src * 128 + lane * 4);
  float p = qv.x * kv.x + qv.y * kv.y + qv.z * kv.z + qv.w * kv.w;
  for (int o = 16; o > 0; o >>= 1) p += __shfl_down(p, o, 32);   // wave32
  if (lane == 0) {
    const float a = p * pHead[0] * 0.08838834764831845f;  // 1/sqrt(128)
    alpha[e] = a;
    atomicMax(&amax[dst], fenc(a));
  }
}

__global__ void edge_ex_kernel(const int* __restrict__ ei, int E,
    float* __restrict__ alpha, const unsigned* __restrict__ amax,
    float* __restrict__ denom)
{
  const int e = blockIdx.x * blockDim.x + threadIdx.x;
  if (e >= E) return;
  const int dst = ei[E + e];
  const float ex = expf(alpha[e] - fdec(amax[dst]));
  alpha[e] = ex;                            // in-place: alpha -> exp
  atomicAdd(&denom[dst], ex);
}

__global__ void edge_scatter_kernel(const int* __restrict__ ei, int E,
    const float* __restrict__ alpha, const float* __restrict__ denom,
    const float* __restrict__ vr, float* __restrict__ aggr)
{
  const int e = blockIdx.x * 8 + (threadIdx.x >> 5);
  if (e >= E) return;
  const int lane = threadIdx.x & 31;
  const int src = ei[e];
  const int dst = ei[E + e];
  const float w = alpha[e] / denom[dst];
  const float4 vv = *(const float4*)(vr + (size_t)src * 128 + lane * 4);
  float* dp = aggr + (size_t)dst * 128 + lane * 4;
  atomicAdd(dp + 0, w * vv.x);
  atomicAdd(dp + 1, w * vv.y);
  atomicAdd(dp + 2, w * vv.z);
  atomicAdd(dp + 3, w * vv.w);
}

// ------------------------------ small kernels ------------------------------
__global__ void fill_kernel(unsigned* __restrict__ p, unsigned v, size_t n) {
  size_t i = (size_t)blockIdx.x * blockDim.x + threadIdx.x;
  if (i < n) p[i] = v;
}

__global__ void numcat_kernel(const float* __restrict__ ux,
    const float* __restrict__ num_w, const float* __restrict__ num_b,
    const float* __restrict__ cat_w, const float* __restrict__ cat_b,
    unsigned short* __restrict__ feat16, int N)
{
  const int idx = blockIdx.x * blockDim.x + threadIdx.x;
  if (idx >= N * 64) return;
  const int n = idx >> 6;
  const int c = idx & 63;
  float acc;
  if (c < 32) {
    acc = num_b[c];
    for (int i = 0; i < 5; ++i) acc += ux[(size_t)n * 1544 + i] * num_w[i * 32 + c];
  } else {
    const int cc = c - 32;
    acc = cat_b[cc];
    for (int i = 0; i < 3; ++i) acc += ux[(size_t)n * 1544 + 5 + i] * cat_w[i * 32 + cc];
  }
  acc = acc > 0.f ? acc : 0.01f * acc;      // leaky_relu
  feat16[(size_t)n * 128 + c] = f2bf(acc);
}

__global__ void head_kernel(const float* __restrict__ x,
    const float* __restrict__ o2w, const float* __restrict__ o2b,
    float* __restrict__ out, int N)
{
  const int n = blockIdx.x * blockDim.x + threadIdx.x;
  if (n >= N) return;
  float a0 = o2b[0], a1 = o2b[1];
  const float* xr = x + (size_t)n * 128;
  for (int i = 0; i < 128; ++i) {
    const float v = xr[i];
    a0 += v * o2w[i * 2 + 0];
    a1 += v * o2w[i * 2 + 1];
  }
  out[n * 2 + 0] = a0;
  out[n * 2 + 1] = a1;
}

// ---------------------------------------------------------------------------
extern "C" void kernel_launch(void* const* d_in, const int* in_sizes, int n_in,
                              void* d_out, int out_size, void* d_ws, size_t ws_size,
                              hipStream_t stream)
{
  const float* user_x    = (const float*)d_in[0];
  const int*   ei_follow = (const int*)  d_in[1];
  const int*   ei_friend = (const int*)  d_in[2];
  const int N = in_sizes[0] / 1544;
  const int E = in_sizes[1] / 2;

  // params dict, insertion order
  const float* num_w = (const float*)d_in[3];
  const float* num_b = (const float*)d_in[4];
  const float* cat_w = (const float*)d_in[5];
  const float* cat_b = (const float*)d_in[6];
  const float* des_w = (const float*)d_in[7];
  const float* des_b = (const float*)d_in[8];
  const float* tw_w  = (const float*)d_in[9];
  const float* tw_b  = (const float*)d_in[10];
  const float* in_w  = (const float*)d_in[11];
  const float* in_b  = (const float*)d_in[12];
  const float* o1_w  = (const float*)d_in[13];
  const float* o1_b  = (const float*)d_in[14];
  const float* o2_w  = (const float*)d_in[15];
  const float* o2_b  = (const float*)d_in[16];

  const size_t NH = (size_t)N * 128;
  char* wp = (char*)d_ws;
  auto carve = [&](size_t bytes) -> char* {
    char* p = wp; wp += (bytes + 255) & ~(size_t)255; return p;
  };
  unsigned short* feat16 = (unsigned short*)carve(NH * 2);
  unsigned short* hA16   = (unsigned short*)carve(NH * 2);
  unsigned short* hB16   = (unsigned short*)carve(NH * 2);
  unsigned short* k16    = (unsigned short*)carve(NH * 2);
  unsigned short* v16    = (unsigned short*)carve(NH * 2);
  float* qf    = (float*)carve(NH * 4);
  float* krf   = (float*)carve(NH * 4);
  float* vrf   = (float*)carve(NH * 4);
  float* aggr  = (float*)carve(NH * 4);
  float* o1out = (float*)carve(NH * 4);
  float* alpha = (float*)carve((size_t)E * 4);
  float* denom = (float*)carve((size_t)N * 4);
  unsigned* amax = (unsigned*)carve((size_t)N * 4);
  (void)ws_size; (void)n_in;

  const dim3 blk(256);
  const dim3 gGemm((N + 127) / 128);
  const dim3 gWaveE((E + 7) / 8);
  const dim3 gE((E + 255) / 256);
  const dim3 gNH((unsigned)((NH + 255) / 256));
  const dim3 gN1((N + 255) / 256);
  const dim3 gNC((N * 64 + 255) / 256);

  // One-shot: convert + swizzle each weight matrix to bf16 fragment order.
  auto prepW = [&](const float* W, int kin, int nout) -> unsigned short* {
    unsigned short* dst = (unsigned short*)carve((size_t)kin * nout * 2);
    const int total = kin * nout;
    wprep_kernel<<<dim3((total + 255) / 256), blk, 0, stream>>>(W, dst, kin, nout);
    return dst;
  };
  unsigned short* w_des = prepW(des_w, 768, 32);
  unsigned short* w_tw  = prepW(tw_w,  768, 32);
  unsigned short* w_in  = prepW(in_w,  128, 128);
  unsigned short* w_o1  = prepW(o1_w,  128, 128);
  unsigned short* w_lay[2][8];   // k,q,v,out, a_f, m_f, a_fr, m_fr
  for (int L = 0; L < 2; ++L) {
    const int b = 17 + 15 * L;
    w_lay[L][0] = prepW((const float*)d_in[b + 0],  128, 128);  // k_w
    w_lay[L][1] = prepW((const float*)d_in[b + 2],  128, 128);  // q_w
    w_lay[L][2] = prepW((const float*)d_in[b + 4],  128, 128);  // v_w
    w_lay[L][3] = prepW((const float*)d_in[b + 6],  128, 128);  // out_w
    w_lay[L][4] = prepW((const float*)d_in[b + 9],  128, 128);  // a_follow
    w_lay[L][5] = prepW((const float*)d_in[b + 10], 128, 128);  // m_follow
    w_lay[L][6] = prepW((const float*)d_in[b + 12], 128, 128);  // a_friend
    w_lay[L][7] = prepW((const float*)d_in[b + 13], 128, 128);  // m_friend
  }

  // -------- feature MLPs -> feat16 (N x 128, bf16) --------
  numcat_kernel<<<gNC, blk, 0, stream>>>(user_x, num_w, num_b, cat_w, cat_b,
                                         feat16, N);
  gemm_wmma_kernel<768, 32, 0, 1, false, false, false, true, true>
      <<<gGemm, blk, 0, stream>>>(user_x, 1544, 8, w_des, des_b,
                                  nullptr, feat16, 128, 64, nullptr, nullptr, N);
  gemm_wmma_kernel<768, 32, 0, 1, false, false, false, true, true>
      <<<gGemm, blk, 0, stream>>>(user_x, 1544, 776, w_tw, tw_b,
                                  nullptr, feat16, 128, 96, nullptr, nullptr, N);

  // -------- input projection: hA16 = leaky(feat @ in_w + in_b) --------
  gemm_wmma_kernel<128, 128, 0, 1, false, true, false, true, true>
      <<<gGemm, blk, 0, stream>>>(feat16, 128, 0, w_in, in_b,
                                  nullptr, hA16, 128, 0, nullptr, nullptr, N);

  // -------- HGT layers --------
  for (int L = 0; L < 2; ++L) {
    const int b = 17 + 15 * L;
    const float* k_b   = (const float*)d_in[b + 1];
    const float* q_b   = (const float*)d_in[b + 3];
    const float* v_b   = (const float*)d_in[b + 5];
    const float* out_b = (const float*)d_in[b + 7];
    const float* skip  = (const float*)d_in[b + 8];
    const float* p_et[2] = { (const float*)d_in[b + 11], (const float*)d_in[b + 14] };
    const int*   ei_et[2] = { ei_follow, ei_friend };

    const unsigned short* hin16  = (L == 0) ? hA16 : hB16;
    unsigned short*       hout16 = (L == 0) ? hB16 : hA16;

    // q (f32, for edge dot), k/v (bf16, feed kr/vr WMMA directly)
    gemm_wmma_kernel<128, 128, 0, 0, false, true, true, false, true>
        <<<gGemm, blk, 0, stream>>>(hin16, 128, 0, w_lay[L][1], q_b,
                                    qf, nullptr, 128, 0, nullptr, nullptr, N);
    gemm_wmma_kernel<128, 128, 0, 0, false, true, false, true, true>
        <<<gGemm, blk, 0, stream>>>(hin16, 128, 0, w_lay[L][0], k_b,
                                    nullptr, k16, 128, 0, nullptr, nullptr, N);
    gemm_wmma_kernel<128, 128, 0, 0, false, true, false, true, true>
        <<<gGemm, blk, 0, stream>>>(hin16, 128, 0, w_lay[L][2], v_b,
                                    nullptr, v16, 128, 0, nullptr, nullptr, N);

    fill_kernel<<<gNH, blk, 0, stream>>>((unsigned*)aggr, 0u, NH);

    for (int et = 0; et < 2; ++et) {
      // kr = k @ a_et, vr = v @ m_et (no bias, f32 out)
      gemm_wmma_kernel<128, 128, 0, 0, false, true, true, false, false>
          <<<gGemm, blk, 0, stream>>>(k16, 128, 0, w_lay[L][4 + 2 * et], nullptr,
                                      krf, nullptr, 128, 0, nullptr, nullptr, N);
      gemm_wmma_kernel<128, 128, 0, 0, false, true, true, false, false>
          <<<gGemm, blk, 0, stream>>>(v16, 128, 0, w_lay[L][5 + 2 * et], nullptr,
                                      vrf, nullptr, 128, 0, nullptr, nullptr, N);

      fill_kernel<<<gN1, blk, 0, stream>>>(amax, 0u, (size_t)N);   // -inf sentinel
      fill_kernel<<<gN1, blk, 0, stream>>>((unsigned*)denom, 0u, (size_t)N);

      edge_alpha_kernel<<<gWaveE, blk, 0, stream>>>(ei_et[et], E, qf, krf,
                                                    p_et[et], alpha, amax);
      edge_ex_kernel<<<gE, blk, 0, stream>>>(ei_et[et], E, alpha, amax, denom);
      edge_scatter_kernel<<<gWaveE, blk, 0, stream>>>(ei_et[et], E, alpha,
                                                      denom, vrf, aggr);
    }

    // hout16 = beta * (gelu(aggr) @ out_w + out_b) + (1-beta) * hin16
    gemm_wmma_kernel<128, 128, 2, 0, true, false, false, true, true>
        <<<gGemm, blk, 0, stream>>>(aggr, 128, 0, w_lay[L][3], out_b,
                                    nullptr, hout16, 128, 0, hin16, skip, N);
  }

  // -------- output head --------
  gemm_wmma_kernel<128, 128, 0, 1, false, true, true, false, true>
      <<<gGemm, blk, 0, stream>>>(hA16, 128, 0, w_o1, o1_b,
                                  o1out, nullptr, 128, 0, nullptr, nullptr, N);
  head_kernel<<<gN1, blk, 0, stream>>>(o1out, o2_w, o2_b, (float*)d_out, N);
  (void)out_size;
}
